// MultiHeadSelfAttention_37185826849161
// MI455X (gfx1250) — matrix operand order repaired and checked by reading, not compiled, not run on hardware
//
#include <hip/hip_runtime.h>
#include <math.h>
#include <stdint.h>

// ---------------------------------------------------------------------------
// MultiHeadSelfAttention for MI455X (gfx1250, wave32, WMMA + TDM).
// Pipeline:  f32->f16 convert  ->  Q/K/V NT-GEMMs (WMMA f16, f32 acc)
//            -> fused flash attention (TDM-staged K/V tiles, WMMA QK^T +
//               online softmax + WMMA PV)
//            -> output projection NT-GEMM (f32 out).
// ---------------------------------------------------------------------------

typedef _Float16 v4h  __attribute__((ext_vector_type(4)));
typedef _Float16 v8h  __attribute__((ext_vector_type(8)));
typedef _Float16 v16h __attribute__((ext_vector_type(16)));
typedef float    v8f  __attribute__((ext_vector_type(8)));
typedef uint32_t su4  __attribute__((ext_vector_type(4)));
typedef uint32_t su8  __attribute__((ext_vector_type(8)));

#define HD    1024   // hidden dim
#define NH    16     // heads
#define HDIM  64     // head dim
#define SEQ   2048
#define BATCH 2

static __device__ __forceinline__ v16h cat8(v8h lo, v8h hi) {
  return __builtin_shufflevector(lo, hi, 0,1,2,3,4,5,6,7,8,9,10,11,12,13,14,15);
}

static __device__ __forceinline__ v8f wmma_f16(v16h a, v16h b, v8f c) {
  // D = A(16x32 f16) * B(32x16 f16) + C(16x16 f32)
  return __builtin_amdgcn_wmma_f32_16x16x32_f16(
      /*neg_a=*/false, a, /*neg_b=*/false, b,
      /*c_mod=*/(short)0, c, /*reuse_a=*/false, /*reuse_b=*/false);
}

// ---------------------------------------------------------------------------
// TDM: DMA a 2D f16 tile (tile_d0 x tile_d1, row pitch stride0 elements) from
// global memory into LDS, adding pad_amount DWORDs after every pad_interval
// DWORDs stored (LDS row padding for bank-conflict avoidance).
// Descriptor layout per CDNA5 ISA 8.3/8.4 (D# group0 = 4 SGPRs, group1 = 8).
// All arguments are workgroup-uniform -> SGPRs.
// ---------------------------------------------------------------------------
static __device__ __forceinline__ void tdm_load_tile_f16(
    uint32_t lds_addr, uint64_t gaddr, uint32_t tile_d0, uint32_t tile_d1,
    uint32_t stride0, uint32_t pad_interval_code, uint32_t pad_amount_code) {
  union { uint32_t u[4]; su4 v; } g0;
  g0.u[0] = 1u;                                   // count=1, no gather
  g0.u[1] = lds_addr;                             // LDS byte address
  g0.u[2] = (uint32_t)gaddr;                      // global addr [31:0]
  g0.u[3] = (uint32_t)((gaddr >> 32) & 0x1FFFFFFu) | (2u << 30);  // type=2

  union { uint64_t q[4]; su8 v; } g1;
  g1.q[0] = (1ull << 16)                              // data_size = 2 bytes
          | (1ull << 20)                              // pad_enable
          | ((uint64_t)pad_interval_code << 22)       // DWORDs between pads
          | ((uint64_t)pad_amount_code << 25)         // pad DWORDs
          | ((uint64_t)(tile_d0 & 0xFFFFu) << 48);    // tensor_dim0[15:0]
  g1.q[1] = ((uint64_t)tile_d0 >> 16)                 // tensor_dim0[31:16]
          | ((uint64_t)tile_d1 << 16)                 // tensor_dim1
          | ((uint64_t)tile_d0 << 48);                // tile_dim0
  g1.q[2] = (uint64_t)tile_d1                         // tile_dim1 (tile_dim2=0)
          | ((uint64_t)stride0 << 32);                // tensor_dim0_stride[31:0]
  g1.q[3] = 0;                                        // stride0[47:32]=0, stride1=0

  asm volatile("tensor_load_to_lds %0, %1" :: "s"(g0.v), "s"(g1.v) : "memory");
}

// ---------------------------------------------------------------------------
// f32 -> f16 conversion, 4 elements per thread iteration.
// ---------------------------------------------------------------------------
__global__ void cvt_f32_to_f16(const float* __restrict__ src,
                               _Float16* __restrict__ dst, int n4) {
  int i = blockIdx.x * blockDim.x + threadIdx.x;
  int stride = gridDim.x * blockDim.x;
  for (; i < n4; i += stride) {
    float4 v = ((const float4*)src)[i];
    v4h h;
    h[0] = (_Float16)v.x; h[1] = (_Float16)v.y;
    h[2] = (_Float16)v.z; h[3] = (_Float16)v.w;
    ((v4h*)dst)[i] = h;
  }
}

// ---------------------------------------------------------------------------
// NT GEMM:  C[M,N] = A[M,Kd] * Bm[N,Kd]^T   (both row-major, K contiguous)
// 128x128 block tile, TK=32, 8 waves (4 along M x 2 along N), wave = 32x64.
// mode 0: f16 row-major C.  mode 1: f16 per-head-transposed (V^T) store.
// mode 2: f32 row-major C.
// ---------------------------------------------------------------------------
__launch_bounds__(256)
__global__ void gemm_nt(const _Float16* __restrict__ A,
                        const _Float16* __restrict__ Bm,
                        void* __restrict__ Cout,
                        int M, int N, int Kd, int mode) {
  __shared__ __align__(16) _Float16 As[128 * 40];  // padded stride: 40 halves
  __shared__ __align__(16) _Float16 Bs[128 * 40];

  const int tid  = threadIdx.x;
  const int lane = tid & 31;
  const int wid  = tid >> 5;
  const int bm   = blockIdx.y * 128;
  const int bn   = blockIdx.x * 128;
  const int waveM = (wid & 3) * 32;
  const int waveN = (wid >> 2) * 64;
  const int l16  = lane & 15;
  const int hiL  = lane >> 4;          // 0 or 1
  const int koff = hiL * 8;            // A-fragment K offset (halves)
  const int boff = hiL * 16;           // unused; B fragments also use koff

  v8f c[2][4] = {};

  for (int k0 = 0; k0 < Kd; k0 += 32) {
    __syncthreads();
    // Stage A/B 128x32 f16 tiles: 512 16B chunks each, 2 per thread.
#pragma unroll
    for (int i = 0; i < 2; ++i) {
      int cidx = tid + i * 256;            // 0..511
      int row  = cidx >> 2;                // 4 chunks of 8 halves per row
      int off  = (cidx & 3) * 8;
      *(v8h*)&As[row * 40 + off] =
          *(const v8h*)(A + (size_t)(bm + row) * Kd + k0 + off);
      *(v8h*)&Bs[row * 40 + off] =
          *(const v8h*)(Bm + (size_t)(bn + row) * Kd + k0 + off);
    }
    __syncthreads();

    v16h a[2], b[4];
#pragma unroll
    for (int t = 0; t < 2; ++t) {
      const _Float16* p = &As[(waveM + t * 16 + l16) * 40];
      a[t] = cat8(*(const v8h*)(p + koff), *(const v8h*)(p + 16 + koff));
    }
#pragma unroll
    for (int nt = 0; nt < 4; ++nt) {
      const _Float16* p = &Bs[(waveN + nt * 16 + l16) * 40];
      b[nt] = cat8(*(const v8h*)(p + koff), *(const v8h*)(p + 16 + koff));
    }
#pragma unroll
    for (int t = 0; t < 2; ++t)
#pragma unroll
      for (int nt = 0; nt < 4; ++nt)
        c[t][nt] = wmma_f16(a[t], b[nt], c[t][nt]);
  }

  // Epilogue per documented C layout: M = vgpr + 8*(lane>=16), N = lane%16.
#pragma unroll
  for (int t = 0; t < 2; ++t) {
#pragma unroll
    for (int v = 0; v < 8; ++v) {
      int m = bm + waveM + t * 16 + v + hiL * 8;
#pragma unroll
      for (int nt = 0; nt < 4; ++nt) {
        int n = bn + waveN + nt * 16 + l16;
        float val = c[t][nt][v];
        if (mode == 2) {
          ((float*)Cout)[(size_t)m * N + n] = val;
        } else if (mode == 0) {
          ((_Float16*)Cout)[(size_t)m * N + n] = (_Float16)val;
        } else {  // mode 1: V^T per head -> Vt[(b*NH+h)*HDIM + d][SEQ] = V
          int bb = m >> 11, s = m & 2047;
          int hh = n >> 6,  d = n & 63;
          ((_Float16*)Cout)[((size_t)((bb * NH + hh) * HDIM + d)) * SEQ + s] =
              (_Float16)val;
        }
      }
    }
  }
}

// ---------------------------------------------------------------------------
// Fused flash attention per (b, h, 64-row q tile). 4 waves x 16 q-rows.
// K/V 64x64 tiles DMA'd into LDS by the Tensor Data Mover (wave 0 issues,
// TENSORcnt + barrier publishes). Online softmax; P staged in LDS row-major
// and re-read in WMMA A-fragment layout (same-wave DS ordering).
// ---------------------------------------------------------------------------
__launch_bounds__(128)
__global__ void attn_fused(const _Float16* __restrict__ Q,
                           const _Float16* __restrict__ K,
                           const _Float16* __restrict__ Vt,
                           const float* __restrict__ mask,
                           _Float16* __restrict__ Y) {
  __shared__ __align__(16) _Float16 Ks[64 * 72];    // K tile  [s_k][hd], padded
  __shared__ __align__(16) _Float16 Vs[64 * 72];    // V^T tile [d][s_k], padded
  __shared__ __align__(16) _Float16 Ps[4][16 * 72]; // per-wave P tile, padded

  const int tid  = threadIdx.x;
  const int lane = tid & 31;
  const int w    = tid >> 5;
  const int l16  = lane & 15;
  const int hiL  = lane >> 4;
  const int koff = hiL * 8;
  const int boff = hiL * 16;
  const int bh   = blockIdx.y;
  const int b    = bh >> 4;
  const int h    = bh & 15;
  const int q0   = blockIdx.x * 64;

  // Q fragments for this wave's 16 rows (A layout, K=64 -> 2 frags).
  const _Float16* qp =
      Q + ((size_t)(b * SEQ + q0 + w * 16 + l16)) * HD + h * HDIM;
  v16h aQ[2];
#pragma unroll
  for (int kk = 0; kk < 2; ++kk)
    aQ[kk] = cat8(*(const v8h*)(qp + kk * 32 + koff),
                  *(const v8h*)(qp + kk * 32 + 16 + koff));

  float m_run[8], l_run[8];
  v8f o[4] = {};
#pragma unroll
  for (int v = 0; v < 8; ++v) { m_run[v] = -INFINITY; l_run[v] = 0.0f; }

  const _Float16* Kg = K  + ((size_t)(b * SEQ)) * HD + h * HDIM;
  const _Float16* Vg = Vt + ((size_t)((b * NH + h) * HDIM)) * SEQ;
  const float*    mk = mask + b * SEQ;

  // Uniform TDM parameters. LDS rows: 64 halves data (32 DW) + 8 halves pad
  // (4 DW)  ->  pad_interval code 4 (=32 DW), pad_amount code 3 (=4 DW).
  const uint32_t ldsK = (uint32_t)(uintptr_t)&Ks[0];
  const uint32_t ldsV = (uint32_t)(uintptr_t)&Vs[0];
  const uint64_t gK = (uint64_t)(uintptr_t)Kg;
  const uint64_t gV = (uint64_t)(uintptr_t)Vg;

  for (int kb = 0; kb < SEQ; kb += 64) {
    __syncthreads();  // all waves finished reading previous K/V tiles
    if (w == 0) {
      // K tile: 64 rows (s_k), 64 cols (hd), row pitch HD elements.
      tdm_load_tile_f16(ldsK, gK + (uint64_t)kb * HD * 2, 64, 64, HD, 4, 3);
      // V^T tile: 64 rows (d), 64 cols (s_k), row pitch SEQ elements.
      tdm_load_tile_f16(ldsV, gV + (uint64_t)kb * 2, 64, 64, SEQ, 4, 3);
      __builtin_amdgcn_s_wait_tensorcnt(0);
    }
    __syncthreads();  // publish TDM-written LDS to all waves

    // Scores: S[16 x 64] = Q_tile * K_tile^T  (B columns = K rows).
    v8f s[4] = {};
#pragma unroll
    for (int kk = 0; kk < 2; ++kk) {
#pragma unroll
      for (int nt = 0; nt < 4; ++nt) {
        const _Float16* p = &Ks[(nt * 16 + l16) * 72 + kk * 32];
        s[nt] = wmma_f16(aQ[kk], cat8(*(const v8h*)(p + koff), *(const v8h*)(p + 16 + koff)),
                         s[nt]);
      }
    }

    float mv[4];
#pragma unroll
    for (int nt = 0; nt < 4; ++nt) mv[nt] = mk[kb + nt * 16 + l16];

    // Online softmax. Row M=v+8*hiL lives across 16 lanes (N=l16).
#pragma unroll
    for (int v = 0; v < 8; ++v) {
      float mx = -INFINITY;
#pragma unroll
      for (int nt = 0; nt < 4; ++nt) {
        float sv = s[nt][v] * 0.03125f + mv[nt];   // 1/sqrt(1024) + mask
        s[nt][v] = sv;
        mx = fmaxf(mx, sv);
      }
      mx = fmaxf(mx, __shfl_xor(mx, 1, 32));
      mx = fmaxf(mx, __shfl_xor(mx, 2, 32));
      mx = fmaxf(mx, __shfl_xor(mx, 4, 32));
      mx = fmaxf(mx, __shfl_xor(mx, 8, 32));
      float mn    = fmaxf(m_run[v], mx);
      float alpha = __expf(m_run[v] - mn);         // first block: exp(-inf)=0
      m_run[v] = mn;
      float rs = 0.0f;
#pragma unroll
      for (int nt = 0; nt < 4; ++nt) {
        float p = __expf(s[nt][v] - mn);
        s[nt][v] = p;
        rs += p;
      }
      rs += __shfl_xor(rs, 1, 32);
      rs += __shfl_xor(rs, 2, 32);
      rs += __shfl_xor(rs, 4, 32);
      rs += __shfl_xor(rs, 8, 32);
      l_run[v] = l_run[v] * alpha + rs;
#pragma unroll
      for (int nt = 0; nt < 4; ++nt) o[nt][v] *= alpha;
    }

    // Spill P (f16) row-major into this wave's LDS tile.
#pragma unroll
    for (int v = 0; v < 8; ++v) {
      int row = v + hiL * 8;
#pragma unroll
      for (int nt = 0; nt < 4; ++nt)
        Ps[w][row * 72 + nt * 16 + l16] = (_Float16)s[nt][v];
    }
    // Same-wave DS ops are in-order; barrier the compiler + drain DScnt.
    asm volatile("s_wait_dscnt 0" ::: "memory");

    // O += P * V  (B columns of V = rows of V^T tile, K contiguous).
#pragma unroll
    for (int kk = 0; kk < 2; ++kk) {
      const _Float16* pp = &Ps[w][l16 * 72 + kk * 32];
      v16h aP = cat8(*(const v8h*)(pp + koff), *(const v8h*)(pp + 16 + koff));
#pragma unroll
      for (int nt = 0; nt < 4; ++nt) {
        const _Float16* p = &Vs[(nt * 16 + l16) * 72 + kk * 32];
        o[nt] = wmma_f16(aP, cat8(*(const v8h*)(p + koff), *(const v8h*)(p + 16 + koff)),
                         o[nt]);
      }
    }
  }

  // Normalize and write Y (heads concatenated: col = h*64 + d), f16.
#pragma unroll
  for (int v = 0; v < 8; ++v) {
    float inv = 1.0f / l_run[v];
    int row = q0 + w * 16 + v + hiL * 8;
#pragma unroll
    for (int nt = 0; nt < 4; ++nt)
      Y[((size_t)(b * SEQ + row)) * HD + h * HDIM + nt * 16 + l16] =
          (_Float16)(o[nt][v] * inv);
  }
}

// ---------------------------------------------------------------------------
extern "C" void kernel_launch(void* const* d_in, const int* in_sizes, int n_in,
                              void* d_out, int out_size, void* d_ws,
                              size_t ws_size, hipStream_t stream) {
  (void)in_sizes; (void)n_in; (void)out_size; (void)ws_size;
  const float* x    = (const float*)d_in[0];
  const float* mask = (const float*)d_in[1];
  const float* Wq   = (const float*)d_in[2];
  const float* Wk   = (const float*)d_in[3];
  const float* Wv   = (const float*)d_in[4];
  const float* Wo   = (const float*)d_in[5];

  char* ws = (char*)d_ws;
  const size_t MB = (size_t)1 << 20;
  _Float16* xh  = (_Float16*)(ws +  0 * MB);  // [4096,1024] f16  (8 MB)
  _Float16* Wqh = (_Float16*)(ws +  8 * MB);  // [1024,1024] f16  (2 MB)
  _Float16* Wkh = (_Float16*)(ws + 10 * MB);
  _Float16* Wvh = (_Float16*)(ws + 12 * MB);
  _Float16* Woh = (_Float16*)(ws + 14 * MB);
  _Float16* Qh  = (_Float16*)(ws + 16 * MB);  // [4096,1024] f16
  _Float16* Kh  = (_Float16*)(ws + 24 * MB);  // [4096,1024] f16
  _Float16* Vth = (_Float16*)(ws + 32 * MB);  // [B*H*64, 2048] f16 (V^T)
  _Float16* Yh  = (_Float16*)(ws + 40 * MB);  // [4096,1024] f16

  cvt_f32_to_f16<<<1024, 256, 0, stream>>>(x,  xh,  (BATCH * SEQ * HD) / 4);
  cvt_f32_to_f16<<< 512, 256, 0, stream>>>(Wq, Wqh, (HD * HD) / 4);
  cvt_f32_to_f16<<< 512, 256, 0, stream>>>(Wk, Wkh, (HD * HD) / 4);
  cvt_f32_to_f16<<< 512, 256, 0, stream>>>(Wv, Wvh, (HD * HD) / 4);
  cvt_f32_to_f16<<< 512, 256, 0, stream>>>(Wo, Woh, (HD * HD) / 4);

  dim3 ggrid(HD / 128, (BATCH * SEQ) / 128);   // (8, 32)
  gemm_nt<<<ggrid, 256, 0, stream>>>(xh, Wqh, Qh,  BATCH * SEQ, HD, HD, 0);
  gemm_nt<<<ggrid, 256, 0, stream>>>(xh, Wkh, Kh,  BATCH * SEQ, HD, HD, 0);
  gemm_nt<<<ggrid, 256, 0, stream>>>(xh, Wvh, Vth, BATCH * SEQ, HD, HD, 1);

  attn_fused<<<dim3(SEQ / 64, BATCH * NH), 128, 0, stream>>>(Qh, Kh, Vth,
                                                             mask, Yh);

  gemm_nt<<<ggrid, 256, 0, stream>>>(Yh, Woh, d_out, BATCH * SEQ, HD, HD, 2);
}
